// MPNN_LSTM_no_skip_46651934769324
// MI455X (gfx1250) — matrix operand order repaired and checked by reading, not compiled, not run on hardware
//
#include <hip/hip_runtime.h>
#include <hip/hip_bf16.h>

typedef __attribute__((ext_vector_type(16))) __bf16 v16bf;
typedef __attribute__((ext_vector_type(8)))  __bf16 v8bf;
typedef __attribute__((ext_vector_type(8)))  float  v8f;

#define WINDOW 16
#define NNODES 10000
#define NFEAT  32
#define NHID   128
#define NTOT   (WINDOW * NNODES)   /* 160000 stacked node rows */
#define BN_EPS 1e-5f

static __device__ __forceinline__ float sigf(float x) {
    return 1.0f / (1.0f + __expf(-x));
}

// ---------------------------------------------------------------- utility ---
__global__ void k_fill(float* __restrict__ p, float v, long n) {
    long i = (long)blockIdx.x * blockDim.x + threadIdx.x;
    if (i < n) p[i] = v;
}

__global__ void k_deg_accum(float* __restrict__ deg, const int* __restrict__ dst,
                            const float* __restrict__ w, int E) {
    int e = blockIdx.x * blockDim.x + threadIdx.x;
    if (e < E) atomicAdd(&deg[dst[e]], w[e]);
}

__global__ void k_rsqrt_inplace(float* __restrict__ p, int n) {
    int i = blockIdx.x * blockDim.x + threadIdx.x;
    if (i < n) p[i] = rsqrtf(p[i]);
}

__global__ void k_bias_comb(const float* __restrict__ a, const float* __restrict__ b,
                            float* __restrict__ o, int n) {
    int i = blockIdx.x * blockDim.x + threadIdx.x;
    if (i < n) o[i] = a[i] + b[i];
}

// -------------------------------------------------------- WMMA bf16 GEMM ---
// C[M,Nc] = A[M,K] * B (+bias, +=C if ACC, ReLU if RELU)
// B row-major [K,Nc] if BT==0, or stored [Nc,K] (logical B = W^T) if BT==1.
// Block tile 128x64 staged through LDS in bf16; 8 waves, each does 16x64 via
// four v_wmma_f32_16x16x32_bf16 per 32-wide K step.
#define BM 128
#define BN 64
#define BK 32
#define A_ROW 40   /* padded k-stride (bf16 elements), 80B rows, 16B aligned */
#define B_ROW 40

template <int BT, int ACC, int RELU>
__global__ __launch_bounds__(256) void k_gemm_wmma(
        const float* __restrict__ A, int lda,
        const float* __restrict__ B, int ldb,
        const float* __restrict__ bias,
        float* __restrict__ C, int ldc,
        int M, int K, int Nc) {
    __shared__ __align__(16) __bf16 As[BM * A_ROW];   // 10240 B
    __shared__ __align__(16) __bf16 Bs[BN * B_ROW];   // 5120 B  (k-major per column)

    int tid  = threadIdx.x;
    int lane = tid & 31, wave = tid >> 5;
    int nblk = Nc / BN;
    int bm   = blockIdx.x / nblk, bn = blockIdx.x - bm * nblk;
    int row0 = bm * BM, col0 = bn * BN;
    int r    = lane & 15;          // row (A) / col (B,C) within 16
    int hi   = lane >> 4;          // lane-group select
    int wrow = wave * 16;          // this wave's row offset inside block tile

    v8f acc[4];
    if (ACC) {
#pragma unroll
        for (int n16 = 0; n16 < 4; ++n16)
#pragma unroll
            for (int v = 0; v < 8; ++v) {
                int row = row0 + wrow + hi * 8 + v;
                acc[n16][v] = (row < M)
                    ? C[(size_t)row * ldc + col0 + n16 * 16 + r] : 0.0f;
            }
    } else {
#pragma unroll
        for (int n16 = 0; n16 < 4; ++n16)
#pragma unroll
            for (int v = 0; v < 8; ++v) acc[n16][v] = 0.0f;
    }

    for (int kb = 0; kb < K; kb += BK) {
        // ---- stage A: 128x32 f32 -> bf16 LDS; thread = (row, 16-float half)
        {
            int arow = tid >> 1, half = tid & 1;
            int g_r  = row0 + arow;
            __bf16* d = As + arow * A_ROW + half * 16;
            if (g_r < M) {
                const float* ap = A + (size_t)g_r * lda + kb + half * 16;
#pragma unroll
                for (int q = 0; q < 16; q += 4) {
                    float4 f = *(const float4*)(ap + q);
                    d[q + 0] = (__bf16)f.x; d[q + 1] = (__bf16)f.y;
                    d[q + 2] = (__bf16)f.z; d[q + 3] = (__bf16)f.w;
                }
            } else {
#pragma unroll
                for (int q = 0; q < 16; ++q) d[q] = (__bf16)0.0f;
            }
        }
        // ---- stage B: 32x64 -> LDS k-major (Bs[n][k])
        if (BT) {   // B stored [Nc,K]: contiguous in k
            int n = tid >> 2, k0 = (tid & 3) * 8;
            const float* bp = B + (size_t)(col0 + n) * ldb + kb + k0;
            __bf16* d = Bs + n * B_ROW + k0;
#pragma unroll
            for (int q = 0; q < 8; q += 4) {
                float4 f = *(const float4*)(bp + q);
                d[q + 0] = (__bf16)f.x; d[q + 1] = (__bf16)f.y;
                d[q + 2] = (__bf16)f.z; d[q + 3] = (__bf16)f.w;
            }
        } else {    // B stored [K,Nc]: contiguous in n, scatter into k-major LDS
            int k = tid >> 3, n0 = (tid & 7) * 8;
            const float* bp = B + (size_t)(kb + k) * ldb + col0 + n0;
#pragma unroll
            for (int q = 0; q < 8; ++q) Bs[(n0 + q) * B_ROW + k] = (__bf16)bp[q];
        }
        __syncthreads();

        // ---- A fragment: two aligned 16B LDS loads per lane
        union { uint4 u; v8bf v; } alo, ahi;
        {
            const __bf16* ap = As + (wrow + r) * A_ROW + hi * 8;
            alo.u = *(const uint4*)ap;
            ahi.u = *(const uint4*)(ap + 16);
        }
        v16bf a = __builtin_shufflevector(alo.v, ahi.v,
                  0, 1, 2, 3, 4, 5, 6, 7, 8, 9, 10, 11, 12, 13, 14, 15);

#pragma unroll
        for (int n16 = 0; n16 < 4; ++n16) {
            union { uint4 u; v8bf v; } blo, bhi;
            const __bf16* bp = Bs + (n16 * 16 + r) * B_ROW + hi * 8;
            blo.u = *(const uint4*)bp;
            bhi.u = *(const uint4*)(bp + 16);
            v16bf b = __builtin_shufflevector(blo.v, bhi.v,
                      0, 1, 2, 3, 4, 5, 6, 7, 8, 9, 10, 11, 12, 13, 14, 15);
            acc[n16] = __builtin_amdgcn_wmma_f32_16x16x32_bf16(
                false, a, false, b, (short)0, acc[n16], false, false);
        }
        __syncthreads();
    }

#pragma unroll
    for (int n16 = 0; n16 < 4; ++n16) {
        int c = col0 + n16 * 16 + r;
        float bb = bias ? bias[c] : 0.0f;
#pragma unroll
        for (int v = 0; v < 8; ++v) {
            int row = row0 + wrow + hi * 8 + v;
            if (row < M) {
                float val = acc[n16][v] + bb;
                if (RELU) val = fmaxf(val, 0.0f);
                C[(size_t)row * ldc + c] = val;
            }
        }
    }
}

// ------------------------------------------------------- GCN edge scatter ---
// One wave32 per edge; each lane handles one float4 (4 of 128 features).
__global__ void k_edge_agg(float* __restrict__ agg, const float* __restrict__ hw,
                           const float* __restrict__ dinv,
                           const int* __restrict__ src, const int* __restrict__ dst,
                           const float* __restrict__ w, int E) {
    int e    = (int)((blockIdx.x * (long)blockDim.x + threadIdx.x) >> 5);
    int lane = threadIdx.x & 31;
    if (e >= E) return;
    int s = src[e], d = dst[e];
    float coef = dinv[s] * w[e] * dinv[d];
    const float4 v = ((const float4*)(hw + (size_t)s * NHID))[lane];
    float* ad = agg + (size_t)d * NHID + lane * 4;
    atomicAdd(&ad[0], coef * v.x);
    atomicAdd(&ad[1], coef * v.y);
    atomicAdd(&ad[2], coef * v.z);
    atomicAdd(&ad[3], coef * v.w);
}

// P = relu(agg + dinv^2 * h + bias); fused per-column sum / sumsq for BN.
#define FIN_ROWS 256
__global__ void k_gcn_finalize(float* __restrict__ P, const float* __restrict__ hw,
                               const float* __restrict__ dinv, const float* __restrict__ bias,
                               float* __restrict__ colsum, float* __restrict__ colsq, int N) {
    int f  = threadIdx.x;                 // 0..127
    int r0 = blockIdx.x * FIN_ROWS;
    int re = min(r0 + FIN_ROWS, N);
    float b = bias[f], s = 0.0f, s2 = 0.0f;
    for (int row = r0; row < re; ++row) {
        float dv = dinv[row];
        size_t idx = (size_t)row * NHID + f;
        float val = fmaxf(P[idx] + dv * dv * hw[idx] + b, 0.0f);
        P[idx] = val;
        s += val; s2 += val * val;
    }
    atomicAdd(&colsum[f], s);
    atomicAdd(&colsq[f], s2);
}

__global__ void k_bn_apply(const float* __restrict__ P, const float* __restrict__ colsum,
                           const float* __restrict__ colsq, const float* __restrict__ g,
                           const float* __restrict__ b, float* __restrict__ out,
                           int ldo, int coff, int N) {
    long i = (long)blockIdx.x * blockDim.x + threadIdx.x;
    if (i >= (long)N * NHID) return;
    int row = (int)(i / NHID), f = (int)(i - (long)row * NHID);
    float inv_n = 1.0f / (float)N;
    float mu  = colsum[f] * inv_n;
    float var = colsq[f] * inv_n - mu * mu;
    float val = (P[i] - mu) * rsqrtf(var + BN_EPS) * g[f] + b[f];
    out[(size_t)row * ldo + coff + f] = val;
}

// --------------------------------------------------------- LSTM pointwise ---
// gates[row, 0:H]=i, [H:2H]=f, [2H:3H]=g, [3H:4H]=o (torch order).
__global__ void k_lstm_pointwise(const float* __restrict__ gates, float* __restrict__ h,
                                 float* __restrict__ c, int n /* rows*H */) {
    int i = blockIdx.x * blockDim.x + threadIdx.x;
    if (i >= n) return;
    int row = i / NHID, j = i - row * NHID;
    const float* gp = gates + (size_t)row * (4 * NHID);
    float gi = sigf(gp[j]);
    float gf = sigf(gp[NHID + j]);
    float gg = tanhf(gp[2 * NHID + j]);
    float go = sigf(gp[3 * NHID + j]);
    float cn = gf * c[i] + gi * gg;
    c[i] = cn;
    h[i] = go * tanhf(cn);
}

// z = [hn1 | hn2 | skip] per node; skip[node][w*NFEAT+f] = x[w*NNODES+node][f]
#define ZC (2 * NHID + WINDOW * NFEAT)    /* 768 */
__global__ void k_build_z(const float* __restrict__ x, const float* __restrict__ h1,
                          const float* __restrict__ h2, float* __restrict__ z) {
    long i = (long)blockIdx.x * blockDim.x + threadIdx.x;
    if (i >= (long)NNODES * ZC) return;
    int row = (int)(i / ZC), col = (int)(i - (long)row * ZC);
    float v;
    if (col < NHID)            v = h1[(size_t)row * NHID + col];
    else if (col < 2 * NHID)   v = h2[(size_t)row * NHID + (col - NHID)];
    else {
        int q = col - 2 * NHID, w = q >> 5, f = q & 31;
        v = x[(size_t)(w * NNODES + row) * NFEAT + f];
    }
    z[i] = v;
}

__global__ void k_fc2(const float* __restrict__ F1, const float* __restrict__ W,
                      const float* __restrict__ b, float* __restrict__ out, int M) {
    int row = blockIdx.x * blockDim.x + threadIdx.x;
    if (row >= M) return;
    const float* fp = F1 + (size_t)row * NHID;
    float acc = b[0];
    for (int k = 0; k < NHID; ++k) acc += fp[k] * W[k];
    out[row] = fmaxf(acc, 0.0f);
}

// -------------------------------------------------------------- launcher ---
static inline void gemm(hipStream_t s, const float* A, int lda, const float* B, int ldb,
                        int bT, const float* bias, float* C, int ldc,
                        int M, int K, int Nc, int acc, int relu) {
    int blocks = ((M + BM - 1) / BM) * (Nc / BN);
    if (bT) {
        if (acc) k_gemm_wmma<1, 1, 0><<<blocks, 256, 0, s>>>(A, lda, B, ldb, bias, C, ldc, M, K, Nc);
        else     k_gemm_wmma<1, 0, 0><<<blocks, 256, 0, s>>>(A, lda, B, ldb, bias, C, ldc, M, K, Nc);
    } else {
        if (relu) k_gemm_wmma<0, 0, 1><<<blocks, 256, 0, s>>>(A, lda, B, ldb, bias, C, ldc, M, K, Nc);
        else      k_gemm_wmma<0, 0, 0><<<blocks, 256, 0, s>>>(A, lda, B, ldb, bias, C, ldc, M, K, Nc);
    }
}

static inline void fill(hipStream_t s, float* p, float v, long n) {
    k_fill<<<(int)((n + 255) / 256), 256, 0, s>>>(p, v, n);
}

extern "C" void kernel_launch(void* const* d_in, const int* in_sizes, int n_in,
                              void* d_out, int out_size, void* d_ws, size_t ws_size,
                              hipStream_t stream) {
    const float* x    = (const float*)d_in[0];
    const int*   src  = (const int*)d_in[1];
    const int    E    = in_sizes[2];
    const int*   dst  = src + E;
    const float* ew   = (const float*)d_in[2];
    const float* W1   = (const float*)d_in[3];
    const float* b1   = (const float*)d_in[4];
    const float* W2   = (const float*)d_in[5];
    const float* b2   = (const float*)d_in[6];
    const float* gbn1 = (const float*)d_in[7];
    const float* bbn1 = (const float*)d_in[8];
    const float* gbn2 = (const float*)d_in[9];
    const float* bbn2 = (const float*)d_in[10];
    const float* Wih1 = (const float*)d_in[11];   // [512,256]
    const float* Whh1 = (const float*)d_in[12];   // [512,128]
    const float* bih1 = (const float*)d_in[13];
    const float* bhh1 = (const float*)d_in[14];
    const float* Wih2 = (const float*)d_in[15];   // [512,128]
    const float* Whh2 = (const float*)d_in[16];   // [512,128]
    const float* bih2 = (const float*)d_in[17];
    const float* bhh2 = (const float*)d_in[18];
    const float* Wfc1 = (const float*)d_in[19];   // [768,128]
    const float* bfc1 = (const float*)d_in[20];
    const float* Wfc2 = (const float*)d_in[21];   // [128,1]
    const float* bfc2 = (const float*)d_in[22];
    float* out = (float*)d_out;

    // ---- workspace layout (floats) ----
    float* ws = (float*)d_ws;
    size_t o = 0;
    float* DINV  = ws + o; o += NTOT;
    float* HW    = ws + o; o += (size_t)NTOT * NHID;
    float* AGG   = ws + o; o += (size_t)NTOT * NHID;          // reused as P
    float* HCAT  = ws + o; o += (size_t)NTOT * 2 * NHID;
    float* GATES = ws + o; o += (size_t)NNODES * 4 * NHID;
    float* HST1  = ws + o; o += (size_t)NNODES * NHID;
    float* CST1  = ws + o; o += (size_t)NNODES * NHID;
    float* HST2  = ws + o; o += (size_t)NNODES * NHID;
    float* CST2  = ws + o; o += (size_t)NNODES * NHID;
    float* COLS  = ws + o; o += NHID;
    float* COLQ  = ws + o; o += NHID;
    float* BIAS1 = ws + o; o += 4 * NHID;
    float* BIAS2 = ws + o; o += 4 * NHID;
    float* Z     = ws + o; o += (size_t)NNODES * ZC;
    float* F1    = ws + o; o += (size_t)NNODES * NHID;
    (void)ws_size; (void)n_in; (void)out_size;

    // ---- degree / normalization ----
    fill(stream, DINV, 1.0f, NTOT);
    k_deg_accum<<<(E + 255) / 256, 256, 0, stream>>>(DINV, dst, ew, E);
    k_rsqrt_inplace<<<(NTOT + 255) / 256, 256, 0, stream>>>(DINV, NTOT);

    // ---- GCN layer 1 ----
    gemm(stream, x, NFEAT, W1, NHID, 0, nullptr, HW, NHID, NTOT, NFEAT, NHID, 0, 0);
    fill(stream, AGG, 0.0f, (long)NTOT * NHID);
    fill(stream, COLS, 0.0f, 2 * NHID);
    k_edge_agg<<<(int)(((long)E * 32 + 255) / 256), 256, 0, stream>>>(AGG, HW, DINV, src, dst, ew, E);
    k_gcn_finalize<<<(NTOT + FIN_ROWS - 1) / FIN_ROWS, NHID, 0, stream>>>(AGG, HW, DINV, b1, COLS, COLQ, NTOT);
    k_bn_apply<<<(int)(((long)NTOT * NHID + 255) / 256), 256, 0, stream>>>(AGG, COLS, COLQ, gbn1, bbn1, HCAT, 2 * NHID, 0, NTOT);

    // ---- GCN layer 2 (input = HCAT cols 0..127) ----
    gemm(stream, HCAT, 2 * NHID, W2, NHID, 0, nullptr, HW, NHID, NTOT, NHID, NHID, 0, 0);
    fill(stream, AGG, 0.0f, (long)NTOT * NHID);
    fill(stream, COLS, 0.0f, 2 * NHID);
    k_edge_agg<<<(int)(((long)E * 32 + 255) / 256), 256, 0, stream>>>(AGG, HW, DINV, src, dst, ew, E);
    k_gcn_finalize<<<(NTOT + FIN_ROWS - 1) / FIN_ROWS, NHID, 0, stream>>>(AGG, HW, DINV, b2, COLS, COLQ, NTOT);
    k_bn_apply<<<(int)(((long)NTOT * NHID + 255) / 256), 256, 0, stream>>>(AGG, COLS, COLQ, gbn2, bbn2, HCAT, 2 * NHID, NHID, NTOT);

    // ---- two stacked LSTMs, interleaved per timestep ----
    k_bias_comb<<<2, 256, 0, stream>>>(bih1, bhh1, BIAS1, 4 * NHID);
    k_bias_comb<<<2, 256, 0, stream>>>(bih2, bhh2, BIAS2, 4 * NHID);
    fill(stream, HST1, 0.0f, (long)4 * NNODES * NHID);   // h1,c1,h2,c2 contiguous

    const int PW = NNODES * NHID;
    for (int t = 0; t < WINDOW; ++t) {
        const float* xt = HCAT + (size_t)t * NNODES * (2 * NHID);
        // layer 1: gates = xt @ Wih1^T + bias ; gates += h1 @ Whh1^T
        gemm(stream, xt, 2 * NHID, Wih1, 2 * NHID, 1, BIAS1, GATES, 4 * NHID, NNODES, 2 * NHID, 4 * NHID, 0, 0);
        gemm(stream, HST1, NHID, Whh1, NHID, 1, nullptr, GATES, 4 * NHID, NNODES, NHID, 4 * NHID, 1, 0);
        k_lstm_pointwise<<<(PW + 255) / 256, 256, 0, stream>>>(GATES, HST1, CST1, PW);
        // layer 2 consumes layer-1 output of this step
        gemm(stream, HST1, NHID, Wih2, NHID, 1, BIAS2, GATES, 4 * NHID, NNODES, NHID, 4 * NHID, 0, 0);
        gemm(stream, HST2, NHID, Whh2, NHID, 1, nullptr, GATES, 4 * NHID, NNODES, NHID, 4 * NHID, 1, 0);
        k_lstm_pointwise<<<(PW + 255) / 256, 256, 0, stream>>>(GATES, HST2, CST2, PW);
    }

    // ---- head: z = [hn1|hn2|skip] ; fc1 relu ; fc2 relu ----
    k_build_z<<<(int)(((long)NNODES * ZC + 255) / 256), 256, 0, stream>>>(x, HST1, HST2, Z);
    gemm(stream, Z, ZC, Wfc1, NHID, 0, bfc1, F1, NHID, NNODES, ZC, NHID, 0, 1);
    k_fc2<<<(NNODES + 255) / 256, 256, 0, stream>>>(F1, Wfc2, bfc2, out, NNODES);
}
